// DSNaiveMoE_20693152432790
// MI455X (gfx1250) — compile-verified, matrix operand order checked by reading, not run on hardware
//
#include <hip/hip_runtime.h>
#include <hip/hip_bf16.h>

// Problem constants (match reference)
#define E_ 8
#define H_ 2048
#define I_ 1408
#define T_ 4096
#define K_ 2
#define NROUTE (T_*K_)      // 8192 total routed rows (sum of per-expert counts)
#define CAP    NROUTE       // worst-case per-expert bucket capacity
#define TM 64               // token tile (rows)
#define TN 128              // output-col tile
#define KT 32               // WMMA K per step (bf16)
#define KTP (KT + 8)        // padded LDS column stride (80B, 16B-aligned, bank-spread)

typedef __attribute__((ext_vector_type(4)))  __bf16 v4bf;
typedef __attribute__((ext_vector_type(8)))  __bf16 v8bf;
typedef __attribute__((ext_vector_type(16))) __bf16 v16bf;
typedef __attribute__((ext_vector_type(8)))  float  v8f;
typedef __attribute__((ext_vector_type(4)))  float  f32x4;
typedef __attribute__((ext_vector_type(4)))  unsigned int u32x4;
typedef __attribute__((ext_vector_type(8)))  int    i32x8;
typedef __attribute__((ext_vector_type(4)))  int    i32x4;

#if defined(__has_builtin)
#if __has_builtin(__builtin_amdgcn_tensor_load_to_lds) && \
    __has_builtin(__builtin_amdgcn_s_wait_tensorcnt)
#define HAVE_TDM 1
#endif
#endif

union Frag16 { v16bf v; v8bf h[2]; };

// Load a 16x32 bf16 fragment from LDS where the "major" index (row for A,
// col for B-transposed) has stride KTP and K is contiguous.
// Layout per ISA 05_wmma: lanes 0-15 -> idx=lane, K {0..7} and {16..23};
// lanes 16-31 -> idx=lane-16, K {8..15} and {24..31}.
__device__ inline v16bf load_frag(const __bf16* base, int lane) {
    int idx   = lane & 15;
    int khalf = (lane >> 4) << 3;          // 0 or 8
    const __bf16* p = base + idx * KTP + khalf;
    Frag16 f;
    f.h[0] = *reinterpret_cast<const v8bf*>(p);        // K khalf..khalf+7
    f.h[1] = *reinterpret_cast<const v8bf*>(p + 16);   // K khalf+16..khalf+23
    return f.v;
}

// Stage a 32(k) x 128(n) f32 tile, transposed+converted into LDS [n][k] bf16.
// Each thread: 4 rows x 4 cols sub-block -> 4x ds_store_b64 (packed v4bf).
__device__ inline void stage_w_tile(const float* __restrict__ src, int row_stride,
                                    __bf16* __restrict__ lds, int tid) {
    const int kq = tid >> 5;          // 0..7 -> k base kq*4
    const int n0 = (tid & 31) * 4;    // 0..124
    const float* s = src + (size_t)(kq * 4) * row_stride + n0;
    f32x4 r0 = *reinterpret_cast<const f32x4*>(s);
    f32x4 r1 = *reinterpret_cast<const f32x4*>(s + row_stride);
    f32x4 r2 = *reinterpret_cast<const f32x4*>(s + 2 * row_stride);
    f32x4 r3 = *reinterpret_cast<const f32x4*>(s + 3 * row_stride);
    #pragma unroll
    for (int j = 0; j < 4; ++j) {
        v4bf t = { (__bf16)r0[j], (__bf16)r1[j], (__bf16)r2[j], (__bf16)r3[j] };
        *reinterpret_cast<v4bf*>(&lds[(n0 + j) * KTP + kq * 4]) = t;
    }
}

// ---------------- routing ----------------

__global__ void moe_route_count(const int* __restrict__ idx, int* __restrict__ counts, int n) {
    int i = blockIdx.x * blockDim.x + threadIdx.x;
    if (i < n) atomicAdd(&counts[idx[i]], 1);
}

__global__ void moe_prefix(const int* __restrict__ counts, int* __restrict__ offsets) {
    if (threadIdx.x == 0 && blockIdx.x == 0) {
        int s = 0;
        for (int e = 0; e < E_; ++e) { offsets[e] = s; s += counts[e]; }
    }
}

__global__ void moe_route_fill(const int* __restrict__ idx, const float* __restrict__ tkw,
                               int* __restrict__ fill, int* __restrict__ bucketTok,
                               float* __restrict__ bucketW, int n) {
    int i = blockIdx.x * blockDim.x + threadIdx.x;
    if (i < n) {
        int e = idx[i];
        int pos = atomicAdd(&fill[e], 1);
        int s = e * CAP + pos;
        bucketTok[s] = i / K_;
        bucketW[s]   = tkw[i];
    }
}

// ---------------- gate/up GEMM + SwiGLU ----------------
// grid: (CAP/TM, I_/TN, E_), block 256 (8 waves)

__global__ __launch_bounds__(256) void moe_gateup(
    const float* __restrict__ X, const float* __restrict__ Wg, const float* __restrict__ Wu,
    const int* __restrict__ counts, const int* __restrict__ offsets,
    const int* __restrict__ bucketTok, __bf16* __restrict__ act)
{
    const int e = blockIdx.z;
    const int count = counts[e];
    const int m0 = blockIdx.x * TM;
    if (m0 >= count) return;                 // block-uniform early exit
    const int i0 = blockIdx.y * TN;
    const int offE = offsets[e];

    __shared__ __align__(16) __bf16 ldsX[TM * KTP];
    __shared__ __align__(16) __bf16 ldsG[TN * KTP];   // transposed: [n][k]
    __shared__ __align__(16) __bf16 ldsU[TN * KTP];   // transposed: [n][k]
    __shared__ int ldsTok[TM];

    const int tid  = threadIdx.x;
    const int lane = tid & 31;
    const int wave = tid >> 5;
    const int rowStrip = wave & 3;           // 4 strips of 16 rows
    const int colHalf  = wave >> 2;          // 0/1 -> 64-col half

    if (tid < TM) {
        int gr = m0 + tid;
        int cr = (gr < count) ? gr : (count - 1);   // clamp (padded rows never stored)
        ldsTok[tid] = bucketTok[e * CAP + cr];
    }
    __syncthreads();

    // hoist loop-invariant addressing out of the K loop
    const int xr  = tid >> 2;
    const int xc0 = (tid & 3) * 8;
    const float* xsrc = X + (size_t)ldsTok[xr] * H_ + xc0;   // advances by KT per step
    const size_t wbase = (size_t)e * H_ * I_;
    const float* sg = Wg + wbase + i0;                       // advances by KT*I_ per step
    const float* su = Wu + wbase + i0;

    v8f accG[4] = {};
    v8f accU[4] = {};

    for (int h0 = 0; h0 < H_; h0 += KT) {
        // stage X tile [64 rows x 32 k] f32->bf16
        {
            f32x4 a = *reinterpret_cast<const f32x4*>(xsrc);
            f32x4 b = *reinterpret_cast<const f32x4*>(xsrc + 4);
            xsrc += KT;
            if (h0 + KT < H_) __builtin_prefetch(xsrc);
            __bf16* dst = &ldsX[xr * KTP + xc0];
            dst[0]=(__bf16)a.x; dst[1]=(__bf16)a.y; dst[2]=(__bf16)a.z; dst[3]=(__bf16)a.w;
            dst[4]=(__bf16)b.x; dst[5]=(__bf16)b.y; dst[6]=(__bf16)b.z; dst[7]=(__bf16)b.w;
        }
        // stage Wg/Wu tiles [32 k x 128 n], transposed into LDS [n][k]
        {
            stage_w_tile(sg, I_, ldsG, tid);
            stage_w_tile(su, I_, ldsU, tid);
            sg += (size_t)KT * I_;
            su += (size_t)KT * I_;
            if (h0 + KT < H_) {
                __builtin_prefetch(sg + (tid & 31) * 4);
                __builtin_prefetch(su + (tid & 31) * 4);
            }
        }
        __syncthreads();

        v16bf a = load_frag(&ldsX[rowStrip * 16 * KTP], lane);
        #pragma unroll
        for (int j = 0; j < 4; ++j) {
            int c = colHalf * 64 + j * 16;
            v16bf bg = load_frag(&ldsG[c * KTP], lane);
            v16bf bu = load_frag(&ldsU[c * KTP], lane);
            accG[j] = __builtin_amdgcn_wmma_f32_16x16x32_bf16(false, a, false, bg,
                                                              (short)0, accG[j], false, false);
            accU[j] = __builtin_amdgcn_wmma_f32_16x16x32_bf16(false, a, false, bu,
                                                              (short)0, accU[j], false, false);
        }
        __syncthreads();
    }

    // epilogue: silu(g)*u -> bf16 act (compact rows: offE + gr)
    const int mbase = rowStrip * 16 + ((lane >> 4) << 3);
    const int ncol  = lane & 15;
    #pragma unroll
    for (int j = 0; j < 4; ++j) {
        int c = i0 + colHalf * 64 + j * 16 + ncol;
        #pragma unroll
        for (int v = 0; v < 8; ++v) {
            int gr = m0 + mbase + v;
            if (gr < count) {
                float g = accG[j][v];
                float u = accU[j][v];
                float s = g * __builtin_amdgcn_rcpf(1.0f + __expf(-g));
                act[(size_t)(offE + gr) * I_ + c] = (__bf16)(s * u);
            }
        }
    }
}

// ---------------- down GEMM + weighted scatter ----------------
// grid: (CAP/TM, H_/TN, E_), block 256
// A-tile (bf16 act rows) staged by the Tensor Data Mover when available:
// one tensor_load_to_lds per K-step issued by wave 0, LDS-padded to KTP stride.

__global__ __launch_bounds__(256) void moe_down(
    const __bf16* __restrict__ act, const float* __restrict__ Wd,
    const int* __restrict__ counts, const int* __restrict__ offsets,
    const int* __restrict__ bucketTok, const float* __restrict__ bucketW,
    float* __restrict__ out)
{
    const int e = blockIdx.z;
    const int count = counts[e];
    const int m0 = blockIdx.x * TM;
    if (m0 >= count) return;
    const int h0 = blockIdx.y * TN;
    const int offE = offsets[e];

    __shared__ __align__(16) __bf16 ldsA[TM * KTP];
    __shared__ __align__(16) __bf16 ldsB[TN * KTP];   // transposed [n][k]
    __shared__ int   ldsTok[TM];
    __shared__ float ldsWt[TM];

    const int tid  = threadIdx.x;
    const int lane = tid & 31;
    const int wave = tid >> 5;
    const int rowStrip = wave & 3;
    const int colHalf  = wave >> 2;

    if (tid < TM) {
        int gr = m0 + tid;
        if (gr < count) { ldsTok[tid] = bucketTok[e*CAP+gr]; ldsWt[tid] = bucketW[e*CAP+gr]; }
        else            { ldsTok[tid] = 0;                   ldsWt[tid] = 0.0f; }
    }
    __syncthreads();

    v8f acc[4] = {};
    const float* sd = Wd + (size_t)e * I_ * H_ + h0;         // advances by KT*H_ per step

#if defined(HAVE_TDM)
    const __bf16* atile = act + (size_t)(offE + m0) * I_;    // tile origin; col advances by KT
    const unsigned ldsA_addr = (unsigned)(unsigned long long)(void*)&ldsA[0]; // flat LSBs = LDS byte addr
    const int rows_left = NROUTE - (offE + m0);              // OOB rows hw-zero-filled
#else
    const int ar  = tid >> 2;
    const int ac0 = (tid & 3) * 8;
    const int agr = m0 + ar;
    const int acr = (agr < count) ? agr : (count - 1);
    const __bf16* asrc = act + (size_t)(offE + acr) * I_ + ac0;  // advances by KT
#endif

    for (int kk0 = 0; kk0 < I_; kk0 += KT) {
#if defined(HAVE_TDM)
        if (wave == 0) {
            // D# group 0: count=1 | lds_addr | global_addr | type=2 ("image")
            unsigned long long ga = (unsigned long long)(const void*)(atile + kk0);
            u32x4 g0;
            g0[0] = 1u;
            g0[1] = ldsA_addr;
            g0[2] = (unsigned)ga;
            g0[3] = (unsigned)((ga >> 32) & 0x01ffffffu) | (2u << 30);
            // D# group 1: data_size=2B; pad +4 DW after every 16 DW (64B row -> 80B stride);
            // tensor extents = remaining cols/rows from tile origin; tile = KT x TM.
            const int dim0 = I_ - kk0;
            i32x8 g1;
            g1[0] = (1 << 16) | (1 << 20) | (3 << 22) | (3 << 25);
            g1[1] = (int)(((unsigned)dim0 & 0xffffu) << 16);            // tensor_dim0 lo16
            g1[2] = (int)(((unsigned)dim0 >> 16) & 0xffffu)             // tensor_dim0 hi16
                  | (int)(((unsigned)rows_left & 0xffffu) << 16);       // tensor_dim1 lo16
            g1[3] = (int)(((unsigned)rows_left >> 16) & 0xffffu)        // tensor_dim1 hi16
                  | (KT << 16);                                         // tile_dim0 = 32
            g1[4] = TM;                                                 // tile_dim1 = 64
            g1[5] = I_;                                                 // tensor_dim0_stride
            g1[6] = 0;
            g1[7] = 0;
            i32x4 z4 = {0, 0, 0, 0};
            i32x8 z8 = {0, 0, 0, 0, 0, 0, 0, 0};
            __builtin_amdgcn_tensor_load_to_lds(g0, g1, z4, z4, z8, 0);
            __builtin_amdgcn_s_wait_tensorcnt((short)0);
        }
#else
        *reinterpret_cast<v8bf*>(&ldsA[ar * KTP + ac0]) =
            *reinterpret_cast<const v8bf*>(asrc);
        asrc += KT;
        if (kk0 + KT < I_) __builtin_prefetch(asrc);
#endif
        // stage Wd tile [32 k x 128 n] transposed, f32->bf16
        stage_w_tile(sd, H_, ldsB, tid);
        sd += (size_t)KT * H_;
        if (kk0 + KT < I_) __builtin_prefetch(sd + (tid & 31) * 4);
        __syncthreads();

        v16bf a = load_frag(&ldsA[rowStrip * 16 * KTP], lane);
        #pragma unroll
        for (int j = 0; j < 4; ++j) {
            v16bf b = load_frag(&ldsB[(colHalf * 64 + j * 16) * KTP], lane);
            acc[j] = __builtin_amdgcn_wmma_f32_16x16x32_bf16(false, a, false, b,
                                                             (short)0, acc[j], false, false);
        }
        __syncthreads();
    }

    // weighted scatter-add into output (each token appears under K_=2 experts)
    const int mbase = rowStrip * 16 + ((lane >> 4) << 3);
    const int ncol  = lane & 15;
    #pragma unroll
    for (int j = 0; j < 4; ++j) {
        int c = h0 + colHalf * 64 + j * 16 + ncol;
        #pragma unroll
        for (int v = 0; v < 8; ++v) {
            int m  = mbase + v;
            int gr = m0 + m;
            if (gr < count) {
                float val = acc[j][v] * ldsWt[m];
                unsafeAtomicAdd(&out[(size_t)ldsTok[m] * H_ + c], val);
            }
        }
    }
}

// ---------------- launcher ----------------

extern "C" void kernel_launch(void* const* d_in, const int* in_sizes, int n_in,
                              void* d_out, int out_size, void* d_ws, size_t ws_size,
                              hipStream_t stream) {
    const float* X   = (const float*)d_in[0];   // [T,H]
    const int*   idx = (const int*)  d_in[1];   // [T,K] (int32 at runtime)
    const float* tkw = (const float*)d_in[2];   // [T,K]
    const float* Wg  = (const float*)d_in[3];   // [E,H,I]
    const float* Wu  = (const float*)d_in[4];   // [E,H,I]
    const float* Wd  = (const float*)d_in[5];   // [E,I,H]
    float* out = (float*)d_out;                 // [T,H]

    char* ws = (char*)d_ws;
    int*    counts    = (int*)(ws + 0);                       // 8 ints
    int*    fill      = (int*)(ws + 32);                      // 8 ints
    int*    offsets   = (int*)(ws + 64);                      // 8 ints
    int*    bucketTok = (int*)(ws + 128);                     // E*CAP ints
    float*  bucketW   = (float*)(ws + 128 + (size_t)E_*CAP*4);// E*CAP floats
    __bf16* act       = (__bf16*)(ws + 128 + (size_t)E_*CAP*8);// NROUTE x I_ bf16

    (void)hipMemsetAsync(ws, 0, 96, stream);                        // counts/fill/offsets
    (void)hipMemsetAsync(d_out, 0, (size_t)T_ * H_ * sizeof(float), stream);

    const int n = NROUTE;
    moe_route_count<<<(n + 255) / 256, 256, 0, stream>>>(idx, counts, n);
    moe_prefix<<<1, 1, 0, stream>>>(counts, offsets);
    moe_route_fill<<<(n + 255) / 256, 256, 0, stream>>>(idx, tkw, fill, bucketTok, bucketW, n);

    dim3 gA(CAP / TM, I_ / TN, E_);   // 128 x 11 x 8, early-exit on count
    moe_gateup<<<gA, 256, 0, stream>>>(X, Wg, Wu, counts, offsets, bucketTok, act);

    dim3 gB(CAP / TM, H_ / TN, E_);   // 128 x 16 x 8
    moe_down<<<gB, 256, 0, stream>>>(act, Wd, counts, offsets, bucketTok, bucketW, out);
}
